// ScaledDotProductAttention_77309411419
// MI455X (gfx1250) — compile-verified, matrix operand order checked by reading, not run on hardware
//
#include <hip/hip_runtime.h>
#include <cstdint>

// ---- problem constants (B=16, S=2048, DK=64, temperature = 8) ----
#define SEQ 2048
#define HD  64
#define BQ  128    // query rows per workgroup
#define NW  8      // waves per workgroup (wave32) -> 2 waves/SIMD
#define NB  16     // batches

typedef __attribute__((ext_vector_type(16))) _Float16 v16h;
typedef __attribute__((ext_vector_type(8)))  float    v8f;
typedef __attribute__((ext_vector_type(2)))  __fp16   fp16x2;  // cvt_pkrtz return type

__device__ __forceinline__ unsigned pk2(float a, float b) {
  fp16x2 t = __builtin_amdgcn_cvt_pkrtz(a, b);   // v_cvt_pk_rtz_f16_f32
  union { fp16x2 h; unsigned u; } c; c.h = t; return c.u;
}

// Gather one 16x32 f16 WMMA operand (A: rows, B: cols) from an LDS row.
// CDNA5 16-bit operand striping: lane L (L%16 = row/col) holds elements
// k = kbase + (L>=16 ? 8 : 0) + (j>=4 ? 16 : 0) + 2*(j&3) .. +1  in VGPR j.
// Two contiguous 16B chunks -> compiler emits 2x ds_load_b128.
__device__ __forceinline__ v16h gather16(const _Float16* rowBase, int kbase, int hi) {
  union { v16h h; unsigned u[8]; } r;
  const unsigned* p = (const unsigned*)rowBase;
  const int b0 = (kbase >> 1) + hi * 4;
  #pragma unroll
  for (int j = 0; j < 8; ++j)
    r.u[j] = p[b0 + ((j >= 4) ? 8 : 0) + (j & 3)];
  return r.h;
}

__device__ __forceinline__ v8f wmma16(v16h a, v16h b, v8f c) {
  return __builtin_amdgcn_wmma_f32_16x16x32_f16(
      /*neg_a=*/false, a, /*neg_b=*/false, b,
      /*c_mod=*/(short)0, c, /*reuse_a=*/false, /*reuse_b=*/false);
}

extern "C" __global__ void __launch_bounds__(256, 1)
sdpa_wmma_kernel(const float* __restrict__ Q, const float* __restrict__ K,
                 const float* __restrict__ V, const uint8_t* __restrict__ Mask,
                 float* __restrict__ Out, float* __restrict__ Attn)
{
  extern __shared__ __align__(16) char smem_raw[];
  _Float16* Kh = (_Float16*)smem_raw;       // [SEQ][HD]  f16   (256 KB)
  _Float16* Qh = Kh + (size_t)SEQ * HD;     // [BQ][HD]   f16   (16 KB)
  _Float16* Vt = Qh + (size_t)BQ * HD;      // per-wave [HD][32] f16 (32 KB)
  _Float16* Pl = Vt + (size_t)NW * HD * 32; // per-wave [16][32] f16 (8 KB)

  const int tid  = threadIdx.x;
  const int w    = tid >> 5;        // wave id 0..7
  const int lane = tid & 31;
  const int ln   = lane & 15;       // N / row-within-16 index
  const int hi   = lane >> 4;       // half-wave select
  const int b    = blockIdx.x >> 4;         // batch (16 q-tiles per batch)
  const int q0   = (blockIdx.x & 15) * BQ;  // query tile base
  const int qw   = q0 + w * 16;             // this wave's first query row

  // ---------- stage K[b] and Q tile (pre-scaled by 1/temperature) ----------
  {
    const float4* ks = (const float4*)(K + (size_t)b * SEQ * HD);
    for (int i = tid; i < SEQ * HD / 4; i += 256) {
      float4 v = ks[i];
      *(uint2*)(Kh + 4 * i) = make_uint2(pk2(v.x, v.y), pk2(v.z, v.w));
    }
    const float4* qs = (const float4*)(Q + ((size_t)b * SEQ + q0) * HD);
    for (int i = tid; i < BQ * HD / 4; i += 256) {
      float4 v = qs[i];   // fold the /8 temperature into Q once
      *(uint2*)(Qh + 4 * i) =
          make_uint2(pk2(v.x * 0.125f, v.y * 0.125f), pk2(v.z * 0.125f, v.w * 0.125f));
    }
  }
  __syncthreads();

  // Q operand (A-matrix layout) held in registers for all passes.
  const _Float16* qrow = Qh + (size_t)(w * 16 + ln) * HD;
  const v16h qa0 = gather16(qrow, 0, hi);
  const v16h qa1 = gather16(qrow, 32, hi);

  // one mask base; row r is at +r*SEQ, tile kt at +kt*16 (lane's column = ln)
  const uint8_t* mbase = Mask + ((size_t)b * SEQ + (size_t)(qw + hi * 8)) * SEQ + ln;

  const float NEG_INF = -__builtin_inff();

  // ---------- pass A: row max + cache mask as register bitmasks ----------
  float mx[8];
  #pragma unroll
  for (int r = 0; r < 8; ++r) mx[r] = NEG_INF;

  unsigned mb[4][8];                         // [word][row] mask bits, reg-resident
  #pragma unroll
  for (int mo = 0; mo < 4; ++mo) {
    unsigned wbits[8];
    #pragma unroll
    for (int r = 0; r < 8; ++r) wbits[r] = 0u;

    #pragma unroll 2
    for (int kt2 = 0; kt2 < 32; ++kt2) {
      const int kt = mo * 32 + kt2;
      const _Float16* krow = Kh + (size_t)(kt * 16 + ln) * HD;
      v16h b0 = gather16(krow, 0, hi);
      v16h b1 = gather16(krow, 32, hi);
      v8f c = {0, 0, 0, 0, 0, 0, 0, 0};
      c = wmma16(qa0, b0, c);
      c = wmma16(qa1, b1, c);
      #pragma unroll
      for (int r = 0; r < 8; ++r) {
        unsigned m8 = mbase[(size_t)r * SEQ + (size_t)kt * 16];
        wbits[r] |= (m8 ? 1u : 0u) << kt2;
        float t = m8 ? NEG_INF : c[r];       // masked_fill(-inf)
        mx[r] = fmaxf(mx[r], t);
      }
    }
    #pragma unroll
    for (int r = 0; r < 8; ++r) mb[mo][r] = wbits[r];
  }
  #pragma unroll
  for (int r = 0; r < 8; ++r) {              // merge across the 16-lane half
    float v = mx[r];
    v = fmaxf(v, __shfl_xor(v, 1, 32));
    v = fmaxf(v, __shfl_xor(v, 2, 32));
    v = fmaxf(v, __shfl_xor(v, 4, 32));
    v = fmaxf(v, __shfl_xor(v, 8, 32));
    mx[r] = v;
  }

  // ---------- pass B: row sum of exp(s - m)  (no global loads) ----------
  float sm[8];
  #pragma unroll
  for (int r = 0; r < 8; ++r) sm[r] = 0.0f;

  #pragma unroll
  for (int mo = 0; mo < 4; ++mo) {
    unsigned wb[8];
    #pragma unroll
    for (int r = 0; r < 8; ++r) wb[r] = mb[mo][r];

    #pragma unroll 2
    for (int kt2 = 0; kt2 < 32; ++kt2) {
      const int kt = mo * 32 + kt2;
      const _Float16* krow = Kh + (size_t)(kt * 16 + ln) * HD;
      v16h b0 = gather16(krow, 0, hi);
      v16h b1 = gather16(krow, 32, hi);
      v8f c = {0, 0, 0, 0, 0, 0, 0, 0};
      c = wmma16(qa0, b0, c);
      c = wmma16(qa1, b1, c);
      #pragma unroll
      for (int r = 0; r < 8; ++r) {
        float t = (wb[r] & 1u) ? NEG_INF : c[r];
        sm[r] += __expf(t - mx[r]);
        wb[r] >>= 1;
      }
    }
  }
  float inv[8];
  #pragma unroll
  for (int r = 0; r < 8; ++r) {
    float v = sm[r];
    v += __shfl_xor(v, 1, 32);
    v += __shfl_xor(v, 2, 32);
    v += __shfl_xor(v, 4, 32);
    v += __shfl_xor(v, 8, 32);
    inv[r] = 1.0f / v;
  }

  // ---------- pass C: emit normalized attn + accumulate O = P @ V ----------
  _Float16* vtw = Vt + (size_t)w * HD * 32;   // wave-private V^T chunk
  _Float16* plw = Pl + (size_t)w * 16 * 32;   // wave-private P tile (A source)
  float* attnB = Attn + (size_t)b * SEQ * SEQ;

  v8f oc[4];
  #pragma unroll
  for (int nt = 0; nt < 4; ++nt) oc[nt] = (v8f){0, 0, 0, 0, 0, 0, 0, 0};

  #pragma unroll
  for (int mo = 0; mo < 4; ++mo) {
    unsigned wb[8];
    #pragma unroll
    for (int r = 0; r < 8; ++r) wb[r] = mb[mo][r];

    for (int kc2 = 0; kc2 < 16; ++kc2) {
      const int kc = mo * 16 + kc2;
      // stage V^T chunk: lane <-> key row, transpose into [dim][32] f16
      {
        const float4* vs = (const float4*)(V + ((size_t)b * SEQ + (size_t)kc * 32 + lane) * HD);
        #pragma unroll
        for (int i = 0; i < HD / 4; ++i) {
          float4 v = vs[i];
          int d = i * 4;
          vtw[(d + 0) * 32 + lane] = (_Float16)v.x;
          vtw[(d + 1) * 32 + lane] = (_Float16)v.y;
          vtw[(d + 2) * 32 + lane] = (_Float16)v.z;
          vtw[(d + 3) * 32 + lane] = (_Float16)v.w;
        }
      }
      if (kc + 1 < SEQ / 32)   // global_prefetch_b8 of next V chunk
        __builtin_prefetch(V + ((size_t)b * SEQ + (size_t)(kc + 1) * 32 + lane) * HD, 0, 0);

      #pragma unroll
      for (int sub = 0; sub < 2; ++sub) {
        const int kt = kc * 2 + sub;
        const _Float16* krow = Kh + (size_t)(kt * 16 + ln) * HD;
        v16h b0 = gather16(krow, 0, hi);
        v16h b1 = gather16(krow, 32, hi);
        v8f c = {0, 0, 0, 0, 0, 0, 0, 0};
        c = wmma16(qa0, b0, c);
        c = wmma16(qa1, b1, c);
        #pragma unroll
        for (int r = 0; r < 8; ++r) {
          const int qr = qw + hi * 8 + r;          // global query row (C layout)
          float t = (wb[r] & 1u) ? NEG_INF : c[r];
          float p = __expf(t - mx[r]) * inv[r];    // normalized probability
          attnB[(size_t)qr * SEQ + (size_t)kt * 16 + ln] = p;     // coalesced b32
          plw[(hi * 8 + r) * 32 + sub * 16 + ln] = (_Float16)p;   // re-layout via LDS
          wb[r] >>= 1;
        }
      }

      // P chunk (16x32) as A, V^T columns as B; contract over 32 keys
      v16h pa = gather16(plw + (size_t)ln * 32, 0, hi);
      #pragma unroll
      for (int nt = 0; nt < 4; ++nt) {
        v16h vb = gather16(vtw + (size_t)(nt * 16 + ln) * 32, 0, hi);
        oc[nt] = wmma16(pa, vb, oc[nt]);
      }
    }
  }

  // ---------- write output (already normalized) ----------
  float* outB = Out + (size_t)b * SEQ * HD;
  #pragma unroll
  for (int nt = 0; nt < 4; ++nt) {
    #pragma unroll
    for (int r = 0; r < 8; ++r) {
      const int qr = qw + hi * 8 + r;
      outB[(size_t)qr * HD + nt * 16 + ln] = oc[nt][r];
    }
  }
}

extern "C" void kernel_launch(void* const* d_in, const int* in_sizes, int n_in,
                              void* d_out, int out_size, void* d_ws, size_t ws_size,
                              hipStream_t stream) {
  (void)in_sizes; (void)n_in; (void)out_size; (void)d_ws; (void)ws_size;
  const float*   q    = (const float*)d_in[0];
  const float*   k    = (const float*)d_in[1];
  const float*   v    = (const float*)d_in[2];
  const uint8_t* mask = (const uint8_t*)d_in[3];

  float* out  = (float*)d_out;                         // [B,S,DK] first
  float* attn = out + (size_t)NB * SEQ * HD;           // then [B,S,S]

  constexpr size_t LDS_BYTES =
      ((size_t)SEQ * HD + (size_t)BQ * HD + (size_t)NW * HD * 32 + (size_t)NW * 16 * 32)
      * sizeof(_Float16);  // 319488 B — fits CDNA5's 320 KB/WGP LDS

  (void)hipFuncSetAttribute((const void*)sdpa_wmma_kernel,
                            hipFuncAttributeMaxDynamicSharedMemorySize,
                            (int)LDS_BYTES);

  const dim3 grid(NB * (SEQ / BQ));   // 16 batches * 16 query tiles = 256
  const dim3 block(NW * 32);          // 8 waves (wave32) -> 2 waves/SIMD
  sdpa_wmma_kernel<<<grid, block, LDS_BYTES, stream>>>(q, k, v, mask, out, attn);
}
